// Attention_33071248179284
// MI455X (gfx1250) — matrix operand order repaired and checked by reading, not compiled, not run on hardware
//
#include <hip/hip_runtime.h>
#include <hip/hip_bf16.h>

// Problem constants (from reference)
#define B_   2
#define T_   2048
#define H_   2048
#define NH_  16
#define HD_  128

typedef __attribute__((ext_vector_type(16))) __bf16 v16bf;
typedef __attribute__((ext_vector_type(8)))  float  v8f;

union AFragU { v16bf v; unsigned short h[16]; };

__device__ __forceinline__ unsigned short f2bf(float f) {
    union { float f; unsigned int u; } c; c.f = f;
    unsigned int u = c.u;
    unsigned int r = u + 0x7FFFu + ((u >> 16) & 1u);   // round-to-nearest-even
    return (unsigned short)(r >> 16);
}

// async DMA: global -> LDS, 16B per lane, tracked by ASYNCcnt
__device__ __forceinline__ void async_b128(unsigned lds_off, unsigned long long gaddr) {
    asm volatile("global_load_async_to_lds_b128 %0, %1, off"
                 :: "v"(lds_off), "v"(gaddr) : "memory");
}
__device__ __forceinline__ void async_b128_o16(unsigned lds_off, unsigned long long gaddr) {
    asm volatile("global_load_async_to_lds_b128 %0, %1, off offset:16"
                 :: "v"(lds_off), "v"(gaddr) : "memory");
}
__device__ __forceinline__ void wait_async0() {
    asm volatile("s_wait_asynccnt 0x0" ::: "memory");
}
__device__ __forceinline__ unsigned lds_off_of(const void* p) {
    return (unsigned)(uintptr_t)p;   // low 32 bits of flat shared addr = LDS offset
}

// ---------------------------------------------------------------------------
// Kernel 1: fp32 -> bf16 conversion, 8 elements / thread
// ---------------------------------------------------------------------------
__global__ void cvt_f32_bf16x8(const float4* __restrict__ in,
                               uint4* __restrict__ out, int n8) {
    int i = blockIdx.x * blockDim.x + threadIdx.x;
    if (i < n8) {
        float4 a = in[i * 2], b = in[i * 2 + 1];
        uint4 o;
        o.x = (unsigned)f2bf(a.x) | ((unsigned)f2bf(a.y) << 16);
        o.y = (unsigned)f2bf(a.z) | ((unsigned)f2bf(a.w) << 16);
        o.z = (unsigned)f2bf(b.x) | ((unsigned)f2bf(b.y) << 16);
        o.w = (unsigned)f2bf(b.z) | ((unsigned)f2bf(b.w) << 16);
        out[i] = o;
    }
}

// ---------------------------------------------------------------------------
// Kernel 2/4: bf16 GEMM  C[M,N] = A[M,K] @ Wt[N,K]^T  via WMMA 16x16x32 bf16.
// 256 thr = 8 waves, tile 64(M) x 128(N), waves 2x4, each 32x32 (2x2 frags).
// K-step 32 with double-buffered LDS filled by async DMA.
// mode 0: fp32 store to Cout.  mode 1: scatter bf16 Q/K (head layout) and
//         V transposed [B,NH,HD,T]; Q scaled by qscale.
// ---------------------------------------------------------------------------
__global__ __launch_bounds__(256) void gemm_bf16_wmma(
    const unsigned short* __restrict__ A,
    const unsigned short* __restrict__ Wt,
    float* __restrict__ Cout,
    unsigned short* __restrict__ Qp,
    unsigned short* __restrict__ Kp,
    unsigned short* __restrict__ Vtp,
    int M, int N, int K, int mode, float qscale)
{
    __shared__ __align__(16) unsigned short sA[2][64 * 40];    // stride 40 (pad 8)
    __shared__ __align__(16) unsigned short sB[2][128 * 40];

    const int tid     = threadIdx.x;
    const int lane    = tid & 31;
    const int wave    = tid >> 5;
    const int wr      = wave >> 2;        // 0..1
    const int wc      = wave & 3;         // 0..3
    const int halfSel = lane >> 4;
    const int l15     = lane & 15;
    const int m0      = blockIdx.y * 64;
    const int n0      = blockIdx.x * 128;

    // staging assignments
    const int ra = tid >> 2, sega = tid & 3;          // A tile 64x32: 1x16B
    const int rb = tid >> 1, segb = (tid & 1) * 2;    // B tile 128x32: 2x16B

    auto issueTiles = [&](int k0, int buf) {
        async_b128(lds_off_of(&sA[buf][ra * 40 + sega * 8]),
                   (unsigned long long)(A + (size_t)(m0 + ra) * K + k0 + sega * 8));
        unsigned ldsB = lds_off_of(&sB[buf][rb * 40 + segb * 8]);
        unsigned long long gB = (unsigned long long)(Wt + (size_t)(n0 + rb) * K + k0 + segb * 8);
        async_b128(ldsB, gB);
        async_b128_o16(ldsB, gB);
        if (k0 + 32 < K) {   // prefetch tile after next
            __builtin_prefetch(A + (size_t)(m0 + ra) * K + k0 + 32 + sega * 8, 0, 2);
            __builtin_prefetch(Wt + (size_t)(n0 + rb) * K + k0 + 32 + segb * 8, 0, 2);
        }
    };

    v8f c[2][2];
#pragma unroll
    for (int i = 0; i < 2; ++i)
#pragma unroll
        for (int j = 0; j < 2; ++j)
#pragma unroll
            for (int r = 0; r < 8; ++r) c[i][j][r] = 0.f;

    const int nk = K >> 5;
    issueTiles(0, 0);
    for (int it = 0; it < nk; ++it) {
        wait_async0();            // own async writes for tile `it` done
        __syncthreads();          // everyone's writes visible
        if (it + 1 < nk) issueTiles((it + 1) << 5, (it + 1) & 1);  // overlap DMA w/ WMMA
        const int buf = it & 1;

        AFragU a[2], b[2];
#pragma unroll
        for (int i = 0; i < 2; ++i) {
            const unsigned short* p = &sA[buf][(wr * 32 + i * 16 + l15) * 40 + halfSel * 16];
#pragma unroll
            for (int g = 0; g < 2; ++g)
#pragma unroll
                for (int j = 0; j < 8; ++j) a[i].h[g * 8 + j] = p[g * 8 + j];
        }
#pragma unroll
        for (int j = 0; j < 2; ++j) {
            const unsigned short* p = &sB[buf][(wc * 32 + j * 16 + l15) * 40 + halfSel * 16];
#pragma unroll
            for (int i = 0; i < 16; ++i) b[j].h[i] = p[i];
        }
#pragma unroll
        for (int i = 0; i < 2; ++i)
#pragma unroll
            for (int j = 0; j < 2; ++j)
                c[i][j] = __builtin_amdgcn_wmma_f32_16x16x32_bf16(
                    false, a[i].v, false, b[j].v, (short)0, c[i][j], false, false);
        __syncthreads();          // all reads of buf done before it is refilled
    }

    const int rowoff = halfSel * 8;
#pragma unroll
    for (int i = 0; i < 2; ++i)
#pragma unroll
        for (int j = 0; j < 2; ++j)
#pragma unroll
            for (int r = 0; r < 8; ++r) {
                int m = m0 + wr * 32 + i * 16 + rowoff + r;
                int n = n0 + wc * 32 + j * 16 + l15;
                float v = c[i][j][r];
                if (mode == 0) {
                    Cout[(size_t)m * N + n] = v;
                } else {
                    int section = n >> 11;          // 0:q 1:k 2:v
                    int hcol = n & 2047;
                    int head = hcol >> 7;
                    int dd   = hcol & 127;
                    int bb   = m >> 11;             // T=2048
                    int tt   = m & 2047;
                    if (section == 0) {
                        size_t dst = (((size_t)(bb * NH_ + head)) * T_ + tt) * HD_ + dd;
                        Qp[dst] = f2bf(v * qscale);
                    } else if (section == 1) {
                        size_t dst = (((size_t)(bb * NH_ + head)) * T_ + tt) * HD_ + dd;
                        Kp[dst] = f2bf(v);
                    } else {                         // V transposed: [B,NH,HD,T]
                        size_t dst = (((size_t)(bb * NH_ + head)) * HD_ + dd) * T_ + tt;
                        Vtp[dst] = f2bf(v);
                    }
                }
            }
}

// ---------------------------------------------------------------------------
// Kernel 3: causal flash attention; Q/K [B,NH,T,HD], V transposed [B,NH,HD,T].
// Grid: (T/128, NH, B). 256 thr = 8 waves; wave w owns 16 q-rows.
// Double-buffered K / Vt tiles filled by async DMA; online softmax in regs.
// ---------------------------------------------------------------------------
__global__ __launch_bounds__(256) void flash_attn_bf16(
    const unsigned short* __restrict__ Q,
    const unsigned short* __restrict__ Kv,
    const unsigned short* __restrict__ Vt,
    unsigned short* __restrict__ Ctx)
{
    __shared__ __align__(16) unsigned short sK[2][32 * 136];   // [key][d], stride 136
    __shared__ __align__(16) unsigned short sVt[2][128 * 40];  // [d][key], stride 40
    __shared__ __align__(16) unsigned short sP[8 * 16 * 32];   // per-wave P

    const int tid     = threadIdx.x;
    const int lane    = tid & 31;
    const int wave    = tid >> 5;
    const int halfSel = lane >> 4;
    const int l15     = lane & 15;
    const int rowoff  = halfSel * 8;

    const int q0 = blockIdx.x * 128;
    const int h  = blockIdx.y;
    const int b  = blockIdx.z;

    const size_t headBase = ((size_t)(b * NH_ + h)) * T_ * HD_;
    const unsigned short* Qh  = Q  + headBase;
    const unsigned short* Kh  = Kv + headBase;
    const unsigned short* Vth = Vt + headBase;      // [HD][T] within head

    const int qw = q0 + wave * 16;

    // staging assignments
    const int rk = tid >> 3, segk = tid & 7;        // K tile 32x128: 1x16B
    const int rv = tid >> 1, segv = (tid & 1) * 2;  // Vt tile 128x32: 2x16B

    auto issueKV = [&](int kv0, int buf) {
        async_b128(lds_off_of(&sK[buf][rk * 136 + segk * 8]),
                   (unsigned long long)(Kh + (size_t)(kv0 + rk) * HD_ + segk * 8));
        unsigned ldsV = lds_off_of(&sVt[buf][rv * 40 + segv * 8]);
        unsigned long long gV = (unsigned long long)(Vth + (size_t)rv * T_ + kv0 + segv * 8);
        async_b128(ldsV, gV);
        async_b128_o16(ldsV, gV);
    };

    // Q fragments (A layout), 4 d-chunks of 32 (softmax scale pre-folded)
    AFragU qf[4];
    {
        const unsigned short* p = Qh + (size_t)(qw + l15) * HD_;
#pragma unroll
        for (int dc = 0; dc < 4; ++dc)
#pragma unroll
            for (int g = 0; g < 2; ++g)
#pragma unroll
                for (int j = 0; j < 8; ++j)
                    qf[dc].h[g * 8 + j] = p[dc * 32 + halfSel * 16 + g * 8 + j];
    }

    float m_r[8], l_r[8];
    v8f o[8];
#pragma unroll
    for (int r = 0; r < 8; ++r) { m_r[r] = -1e30f; l_r[r] = 0.f; }
#pragma unroll
    for (int f = 0; f < 8; ++f)
#pragma unroll
        for (int r = 0; r < 8; ++r) o[f][r] = 0.f;

    const int kv_end = q0 + 128;   // causal upper bound for this block
    issueKV(0, 0);
    for (int kv0 = 0; kv0 < kv_end; kv0 += 32) {
        wait_async0();
        __syncthreads();
        if (kv0 + 32 < kv_end) issueKV(kv0 + 32, ((kv0 >> 5) + 1) & 1);
        const int buf = (kv0 >> 5) & 1;

        // S = Q @ K^T  (two 16-key subtiles x 4 d-chunks)
        v8f s[2];
#pragma unroll
        for (int kt = 0; kt < 2; ++kt) {
#pragma unroll
            for (int r = 0; r < 8; ++r) s[kt][r] = 0.f;
#pragma unroll
            for (int dc = 0; dc < 4; ++dc) {
                AFragU bk;
                const unsigned short* p = &sK[buf][(kt * 16 + l15) * 136 + dc * 32 + halfSel * 16];
#pragma unroll
                for (int i = 0; i < 16; ++i) bk.h[i] = p[i];
                s[kt] = __builtin_amdgcn_wmma_f32_16x16x32_bf16(
                    false, qf[dc].v, false, bk.v, (short)0, s[kt], false, false);
            }
        }

        // causal mask
#pragma unroll
        for (int kt = 0; kt < 2; ++kt)
#pragma unroll
            for (int r = 0; r < 8; ++r) {
                int qrow = qw + rowoff + r;
                int key  = kv0 + kt * 16 + l15;
                if (key > qrow) s[kt][r] = -1e30f;
            }

        // row max across the 16 lanes holding a row
        float mx[8];
#pragma unroll
        for (int r = 0; r < 8; ++r) mx[r] = fmaxf(s[0][r], s[1][r]);
#pragma unroll
        for (int off = 1; off < 16; off <<= 1)
#pragma unroll
            for (int r = 0; r < 8; ++r)
                mx[r] = fmaxf(mx[r], __shfl_xor(mx[r], off, 32));

        float m_new[8], alpha[8];
#pragma unroll
        for (int r = 0; r < 8; ++r) {
            m_new[r] = fmaxf(m_r[r], mx[r]);
            alpha[r] = __expf(m_r[r] - m_new[r]);
        }

        // probs + row sums
        float ps[8];
#pragma unroll
        for (int r = 0; r < 8; ++r) ps[r] = 0.f;
#pragma unroll
        for (int kt = 0; kt < 2; ++kt)
#pragma unroll
            for (int r = 0; r < 8; ++r) {
                float p = __expf(s[kt][r] - m_new[r]);
                s[kt][r] = p;
                ps[r] += p;
            }
#pragma unroll
        for (int off = 1; off < 16; off <<= 1)
#pragma unroll
            for (int r = 0; r < 8; ++r)
                ps[r] += __shfl_xor(ps[r], off, 32);

#pragma unroll
        for (int r = 0; r < 8; ++r) {
            l_r[r] = l_r[r] * alpha[r] + ps[r];
            m_r[r] = m_new[r];
        }

        // P (C layout) -> per-wave LDS -> A layout
        {
            unsigned short* pw = &sP[wave * 512];
#pragma unroll
            for (int kt = 0; kt < 2; ++kt)
#pragma unroll
                for (int r = 0; r < 8; ++r)
                    pw[(rowoff + r) * 32 + kt * 16 + l15] = f2bf(s[kt][r]);
        }
        asm volatile("s_wait_dscnt 0x0" ::: "memory");  // wave-private LDS RAW

        // rescale accumulator
#pragma unroll
        for (int f = 0; f < 8; ++f)
#pragma unroll
            for (int r = 0; r < 8; ++r) o[f][r] *= alpha[r];

        AFragU pa;
        {
            const unsigned short* pw = &sP[wave * 512 + l15 * 32];
            int kb = halfSel * 16;
#pragma unroll
            for (int g = 0; g < 2; ++g)
#pragma unroll
                for (int j = 0; j < 8; ++j) pa.h[g * 8 + j] = pw[g * 8 + kb + j];
        }

        // O += P @ V : B frag contiguous from transposed V tile
#pragma unroll
        for (int nd = 0; nd < 8; ++nd) {
            AFragU bv;
            const unsigned short* p = &sVt[buf][(nd * 16 + l15) * 40 + halfSel * 16];
#pragma unroll
            for (int i = 0; i < 16; ++i) bv.h[i] = p[i];
            o[nd] = __builtin_amdgcn_wmma_f32_16x16x32_bf16(
                false, pa.v, false, bv.v, (short)0, o[nd], false, false);
        }
        __syncthreads();   // all reads of buf done before refill
    }

    // normalize and store context (bf16, [B*T, H] row-major)
#pragma unroll
    for (int nd = 0; nd < 8; ++nd)
#pragma unroll
        for (int r = 0; r < 8; ++r) {
            int qrow = qw + rowoff + r;
            float v = o[nd][r] / l_r[r];
            size_t dst = ((size_t)(b * T_ + qrow)) * H_ + h * HD_ + nd * 16 + l15;
            Ctx[dst] = f2bf(v);
        }
}

// ---------------------------------------------------------------------------
// Host launcher
// ---------------------------------------------------------------------------
extern "C" void kernel_launch(void* const* d_in, const int* in_sizes, int n_in,
                              void* d_out, int out_size, void* d_ws, size_t ws_size,
                              hipStream_t stream) {
    (void)in_sizes; (void)n_in; (void)out_size;

    const float* hs   = (const float*)d_in[0];   // [B,T,H]
    const float* wqkv = (const float*)d_in[1];   // [3H,H]
    const float* wout = (const float*)d_in[2];   // [H,H]
    float* out = (float*)d_out;

    const size_t nX    = (size_t)B_ * T_ * H_;        // 8388608
    const size_t nWqkv = (size_t)3 * H_ * H_;         // 12582912
    const size_t nWout = (size_t)H_ * H_;             // 4194304
    const size_t nQKV  = (size_t)B_ * NH_ * T_ * HD_; // 8388608

    char* ws = (char*)d_ws;
    size_t off = 0;
    auto carve = [&](size_t bytes) -> unsigned short* {
        unsigned short* p = (unsigned short*)(ws + off);
        off = (off + bytes + 255) & ~(size_t)255;
        return p;
    };
    unsigned short* Xbf    = carve(nX * 2);      // reused as Ctx after QKV GEMM
    unsigned short* Wqkvbf = carve(nWqkv * 2);
    unsigned short* Woutbf = carve(nWout * 2);
    unsigned short* Qb     = carve(nQKV * 2);
    unsigned short* Kb     = carve(nQKV * 2);
    unsigned short* Vtb    = carve(nQKV * 2);    // transposed V
    if (off > ws_size) return;  // deterministic no-op if ws too small

    // 1) fp32 -> bf16 (8 elems/thread)
    cvt_f32_bf16x8<<<(int)(nX    / 8 / 256), 256, 0, stream>>>((const float4*)hs,   (uint4*)Xbf,    (int)(nX / 8));
    cvt_f32_bf16x8<<<(int)(nWqkv / 8 / 256), 256, 0, stream>>>((const float4*)wqkv, (uint4*)Wqkvbf, (int)(nWqkv / 8));
    cvt_f32_bf16x8<<<(int)(nWout / 8 / 256), 256, 0, stream>>>((const float4*)wout, (uint4*)Woutbf, (int)(nWout / 8));

    // 2) QKV projection; Q pre-scaled by 1/sqrt(HD); V stored transposed
    {
        dim3 grid(3 * H_ / 128, (B_ * T_) / 64);
        gemm_bf16_wmma<<<grid, 256, 0, stream>>>(
            Xbf, Wqkvbf, nullptr, Qb, Kb, Vtb,
            B_ * T_, 3 * H_, H_, 1, 0.08838834764831845f);
    }

    // 3) causal flash attention -> Ctx (reuse Xbf buffer)
    {
        dim3 grid(T_ / 128, NH_, B_);
        flash_attn_bf16<<<grid, 256, 0, stream>>>(Qb, Kb, Vtb, Xbf);
    }

    // 4) output projection -> fp32 d_out
    {
        dim3 grid(H_ / 128, (B_ * T_) / 64);
        gemm_bf16_wmma<<<grid, 256, 0, stream>>>(
            Xbf, Woutbf, out, nullptr, nullptr, nullptr,
            B_ * T_, H_, H_, 0, 1.0f);
    }
}
